// SVDHead_46651934769529
// MI455X (gfx1250) — compile-verified, matrix operand order checked by reading, not run on hardware
//
#include <hip/hip_runtime.h>
#include <math.h>

#define BATCH 16
#define DIM   512
#define NPTS  1024
#define KNN   10
#define NPAIR 45
#define EPSV  1e-6f

typedef __attribute__((ext_vector_type(16))) __bf16       v16bf;
typedef __attribute__((ext_vector_type(8)))  float        v8f;
typedef __attribute__((ext_vector_type(4)))  unsigned int u32x4;

// ---------------------------------------------------------------------------
// float -> bf16 (round-to-nearest-even), pure integer math (no __bf16 arith)
// ---------------------------------------------------------------------------
__device__ inline unsigned short f2bf_rne(float f) {
  unsigned int u = __float_as_uint(f);
  unsigned int r = u + 0x7FFFu + ((u >> 16) & 1u);
  return (unsigned short)(r >> 16);
}

// ---------------------------------------------------------------------------
// Kernel 0: (B, D, N) fp32  ->  (B, N, D) bf16 hi + bf16 lo  (split precision)
// LDS-tiled 32x32 transpose so both global read and write are coalesced.
// ---------------------------------------------------------------------------
__global__ __launch_bounds__(256) void split_transpose_kernel(
    const float* __restrict__ in, unsigned short* __restrict__ hi,
    unsigned short* __restrict__ lo) {
  __shared__ float tile[32][33];
  const int b  = blockIdx.x;
  const int n0 = blockIdx.y * 32;
  const int d0 = blockIdx.z * 32;
  const int tx = threadIdx.x, ty = threadIdx.y;
  const float* src = in + (size_t)b * DIM * NPTS;
#pragma unroll
  for (int i = 0; i < 4; ++i)
    tile[ty + 8 * i][tx] = src[(size_t)(d0 + ty + 8 * i) * NPTS + (n0 + tx)];
  __syncthreads();
  unsigned short* ho = hi + (size_t)b * NPTS * DIM;
  unsigned short* lout = lo + (size_t)b * NPTS * DIM;
#pragma unroll
  for (int i = 0; i < 4; ++i) {
    float v = tile[tx][ty + 8 * i];
    unsigned short h = f2bf_rne(v);
    float hf = __uint_as_float((unsigned int)h << 16);
    unsigned short l = f2bf_rne(v - hf);
    size_t o = (size_t)(n0 + ty + 8 * i) * DIM + (d0 + tx);
    ho[o] = h;
    lout[o] = l;
  }
}

// ---------------------------------------------------------------------------
// WMMA 16-bit A/B fragment loader (16x32 MxK / 32x16 KxN).
// ISA layout: lanes 0-15 hold K={k..k+7, k+16..k+23}, lanes 16-31 hold
// K={k+8..k+15, k+24..k+31} for row M (A) / col N (B) = lane & 15.
// With row-major NxD bf16 data this is two contiguous 16B loads per lane.
// ---------------------------------------------------------------------------
__device__ inline v16bf load_frag(const unsigned short* p) {
  v16bf f;
  ((u32x4*)&f)[0] = *(const u32x4*)(p);
  ((u32x4*)&f)[1] = *(const u32x4*)(p + 16);
  return f;
}

// ---------------------------------------------------------------------------
// Kernel 1: fused  scores = A^T B / sqrt(512)  ->  softmax(1e4 * s) ->
//           argmax (corres), p @ tgt^T (corr_tgt).   Flash-attention style:
//           never materializes the 16 x 1024 x 1024 score tensor.
// Block = 1 batch x 16-row tile, 4 wave32s; wave w owns m-tiles w, w+4, ...
// 3xBF16 split WMMA: hi*hi + hi*lo + lo*hi accumulated in f32.  (Compiler
// hoists all A fragments into persistent VGPRs across the unrolled k-loop.)
// ---------------------------------------------------------------------------
__global__ __launch_bounds__(128) void attn_kernel(
    const unsigned short* __restrict__ Ah, const unsigned short* __restrict__ Al,
    const unsigned short* __restrict__ Bh, const unsigned short* __restrict__ Bl,
    const float* __restrict__ tgt, float* __restrict__ corr_tgt,
    int* __restrict__ corres_i, float* __restrict__ corres_f) {
  __shared__ float sc[4][16][16];
  __shared__ float red[4][16][5];
  __shared__ float redv[4][16];
  __shared__ int   redi[4][16];

  const int b    = blockIdx.x >> 6;
  const int n0   = (blockIdx.x & 63) << 4;
  const int w    = threadIdx.x >> 5;
  const int lane = threadIdx.x & 31;
  const int lr   = lane & 15;           // A row / B col within tile
  const int kh   = (lane >> 4) << 3;    // K half select (0 or 8)
  const float SCALE = 441.94173824159216f;  // 10000 / sqrt(512)

  const size_t eoff = (size_t)b * NPTS * DIM;
  const unsigned short* aph = Ah + eoff + (size_t)(n0 + lr) * DIM + kh;
  const unsigned short* apl = Al + eoff + (size_t)(n0 + lr) * DIM + kh;
  const float* tg = tgt + (size_t)b * 3 * NPTS;

  float runMax = -INFINITY, runSum = 0.f, vx = 0.f, vy = 0.f, vz = 0.f;
  float amaxV = -INFINITY;
  int   amaxI = 0;

  for (int j = w; j < 64; j += 4) {     // 16 m-tiles per wave, uniform count
    const int m0 = j << 4;
    const unsigned short* bph = Bh + eoff + (size_t)(m0 + lr) * DIM + kh;
    const unsigned short* bpl = Bl + eoff + (size_t)(m0 + lr) * DIM + kh;
    v8f acc = {};
    for (int kk = 0; kk < DIM; kk += 32) {
      v16bf a_h = load_frag(aph + kk);
      v16bf a_l = load_frag(apl + kk);
      v16bf b_h = load_frag(bph + kk);
      v16bf b_l = load_frag(bpl + kk);
      acc = __builtin_amdgcn_wmma_f32_16x16x32_bf16(false, a_h, false, b_h,
                                                    (short)0, acc, false, false);
      acc = __builtin_amdgcn_wmma_f32_16x16x32_bf16(false, a_h, false, b_l,
                                                    (short)0, acc, false, false);
      acc = __builtin_amdgcn_wmma_f32_16x16x32_bf16(false, a_l, false, b_h,
                                                    (short)0, acc, false, false);
      __builtin_prefetch(bph + kk + 256, 0, 3);   // global_prefetch_b8
    }
    // C layout: VGPR v, lanes 0-15 -> M=v, N=lane; lanes 16-31 -> M=v+8.
#pragma unroll
    for (int v = 0; v < 8; ++v) sc[w][v + kh][lr] = acc[v];
    __syncthreads();
    if (lane < 16) {                    // lane r owns row r: online softmax
#pragma unroll
      for (int c = 0; c < 16; ++c) {
        float val = sc[w][lane][c] * SCALE;
        int m = m0 + c;
        if (val > amaxV) { amaxV = val; amaxI = m; }
        if (val > runMax) {
          float f = __expf(runMax - val);
          runSum *= f; vx *= f; vy *= f; vz *= f;
          runMax = val;
        }
        float p = __expf(val - runMax);
        runSum += p;
        vx += p * tg[m];
        vy += p * tg[NPTS + m];
        vz += p * tg[2 * NPTS + m];
      }
    }
    __syncthreads();
  }

  if (lane < 16) {
    red[w][lane][0] = runMax; red[w][lane][1] = runSum;
    red[w][lane][2] = vx;     red[w][lane][3] = vy;    red[w][lane][4] = vz;
    redv[w][lane] = amaxV;    redi[w][lane] = amaxI;
  }
  __syncthreads();
  if (threadIdx.x < 16) {               // merge the 4 wave partials per row
    const int r = threadIdx.x;
    float M = -INFINITY, S = 0.f, X = 0.f, Y = 0.f, Z = 0.f;
    float aV = -INFINITY;
    int   aI = 0x7fffffff;
#pragma unroll
    for (int ww = 0; ww < 4; ++ww) {
      float mw = red[ww][r][0];
      float nm = fmaxf(M, mw);
      float f0 = __expf(M - nm), f1 = __expf(mw - nm);
      S = S * f0 + red[ww][r][1] * f1;
      X = X * f0 + red[ww][r][2] * f1;
      Y = Y * f0 + red[ww][r][3] * f1;
      Z = Z * f0 + red[ww][r][4] * f1;
      M = nm;
      float av = redv[ww][r]; int ai = redi[ww][r];
      if (av > aV || (av == aV && ai < aI)) { aV = av; aI = ai; }
    }
    float inv = 1.f / S;
    size_t o = (size_t)b * NPTS + (n0 + r);
    corr_tgt[o * 3 + 0] = X * inv;
    corr_tgt[o * 3 + 1] = Y * inv;
    corr_tgt[o * 3 + 2] = Z * inv;
    corres_i[o] = aI;
    corres_f[o] = (float)aI;
  }
}

// ---------------------------------------------------------------------------
// Kernel 2: _gfm per point: KNN(10) on src, 45 triangle-area losses against
// corr_tgt, two 45-element sorts (sorted_loss, median), mean of sqrt of
// first 10 -> mean_loss.
// The 2 x 12 KB contiguous point blocks are staged into LDS with the CDNA5
// async global->LDS engine (GLOBAL_LOAD_ASYNC_TO_LDS_B128, ASYNCcnt): no
// VGPR round-trip.  Inline asm is used since the clang builtin arity differs
// across toolchains.
// ---------------------------------------------------------------------------
__global__ __launch_bounds__(256) void gfm_kernel(
    const float* __restrict__ src, const float* __restrict__ corr_tgt,
    float* __restrict__ mloss) {
  __shared__ __attribute__((aligned(16))) float spts[3 * NPTS];  // [c][n]
  __shared__ __attribute__((aligned(16))) float cpts[3 * NPTS];  // [n][c]
  const int b = blockIdx.x >> 2;
  const int n = ((blockIdx.x & 3) << 8) + threadIdx.x;
  const float* sp = src + (size_t)b * 3 * NPTS;        // 12 KB contiguous
  const float* cp = corr_tgt + (size_t)b * NPTS * 3;   // 12 KB contiguous

  {
    const unsigned sbase = (unsigned)(size_t)&spts[0];
    const unsigned cbase = (unsigned)(size_t)&cpts[0];
    for (int ch = threadIdx.x; ch < (3 * NPTS) / 4; ch += 256) {  // 16B chunks
      const float* ga = sp + ch * 4;
      unsigned la = sbase + ch * 16;
      asm volatile("global_load_async_to_lds_b128 %0, %1, off"
                   :: "v"(la), "v"(ga) : "memory");
      const float* gb = cp + ch * 4;
      unsigned lb = cbase + ch * 16;
      asm volatile("global_load_async_to_lds_b128 %0, %1, off"
                   :: "v"(lb), "v"(gb) : "memory");
    }
    asm volatile("s_wait_asynccnt 0x0" ::: "memory");  // this wave's DMAs done
  }
  __syncthreads();                                     // all waves' DMAs done

  const float* sx = &spts[0];
  const float* sy = &spts[NPTS];
  const float* sz = &spts[2 * NPTS];

  float kd[KNN]; int ki[KNN];
#pragma unroll
  for (int i = 0; i < KNN; ++i) { kd[i] = INFINITY; ki[i] = 0; }
  const float px = sx[n], py = sy[n], pz = sz[n];
  for (int m = 0; m < NPTS; ++m) {        // stable top-10 smallest d2
    float dx = px - sx[m], dy = py - sy[m], dz = pz - sz[m];
    float d2 = dx * dx + dy * dy + dz * dz;
    if (d2 < kd[KNN - 1]) {
      int p = KNN - 1;
      while (p > 0 && kd[p - 1] > d2) { kd[p] = kd[p - 1]; ki[p] = ki[p - 1]; --p; }
      kd[p] = d2; ki[p] = m;
    }
  }

  float loss[NPAIR], reg[NPAIR], total[NPAIR], tmp[NPAIR];
  const float qx = cpts[3 * n + 0], qy = cpts[3 * n + 1], qz = cpts[3 * n + 2];
  int p = 0;
  for (int i = 0; i < KNN - 1; ++i) {
    const int ii = ki[i];
    const float e1x = sx[ii] - px, e1y = sy[ii] - py, e1z = sz[ii] - pz;
    const float f1x = cpts[3 * ii + 0] - qx;
    const float f1y = cpts[3 * ii + 1] - qy;
    const float f1z = cpts[3 * ii + 2] - qz;
    for (int j = i + 1; j < KNN; ++j) {
      const int jj = ki[j];
      float e2x = sx[jj] - px, e2y = sy[jj] - py, e2z = sz[jj] - pz;
      float nx = e1y * e2z - e1z * e2y;
      float ny = e1z * e2x - e1x * e2z;
      float nz = e1x * e2y - e1y * e2x;
      float a_s = 0.5f * sqrtf(nx * nx + ny * ny + nz * nz);
      float f2x = cpts[3 * jj + 0] - qx;
      float f2y = cpts[3 * jj + 1] - qy;
      float f2z = cpts[3 * jj + 2] - qz;
      float gx = f1y * f2z - f1z * f2y;
      float gy = f1z * f2x - f1x * f2z;
      float gz = f1x * f2y - f1y * f2x;
      float a_t = 0.5f * sqrtf(gx * gx + gy * gy + gz * gz);
      // src_len=(0,0,a_s), tgt_len=(eps,eps,a_t+eps)
      float d = a_s - a_t - EPSV;
      float num = 2.f * EPSV * EPSV + d * d;
      loss[p] = num / (a_s + a_t + 3.f * EPSV);
      reg[p]  = sqrtf(num);
      ++p;
    }
  }
  for (int a = 1; a < NPAIR; ++a) {       // sort loss ascending
    float key = loss[a]; int t = a - 1;
    while (t >= 0 && loss[t] > key) { loss[t + 1] = loss[t]; --t; }
    loss[t + 1] = key;
  }
  for (int q = 0; q < NPAIR; ++q) { total[q] = loss[q] + 0.1f * reg[q]; tmp[q] = total[q]; }
  for (int a = 1; a < NPAIR; ++a) {       // sort copy for median (45 odd -> [22])
    float key = tmp[a]; int t = a - 1;
    while (t >= 0 && tmp[t] > key) { tmp[t + 1] = tmp[t]; --t; }
    tmp[t + 1] = key;
  }
  const float thr = 3.f * tmp[22];
  float s = 0.f;
  for (int q = 0; q < KNN; ++q) {
    float t = total[q];
    t = (t > thr) ? 0.f : t;
    s += sqrtf(t + EPSV);
  }
  mloss[(size_t)b * NPTS + n] = s * (1.f / KNN);
}

// ---------------------------------------------------------------------------
// Kernel 3: per-batch min(mean_loss) + binary weight = (2*sigmoid(-20*(ml-min)) > 0.5)
// ---------------------------------------------------------------------------
__global__ __launch_bounds__(256) void weight_kernel(
    const float* __restrict__ mloss, float* __restrict__ wout) {
  __shared__ float rmin[256];
  const int b = blockIdx.x;
  const float* ml = mloss + (size_t)b * NPTS;
  float mn = INFINITY;
  for (int n = threadIdx.x; n < NPTS; n += 256) mn = fminf(mn, ml[n]);
  rmin[threadIdx.x] = mn;
  __syncthreads();
  for (int st = 128; st > 0; st >>= 1) {
    if (threadIdx.x < st) rmin[threadIdx.x] = fminf(rmin[threadIdx.x], rmin[threadIdx.x + st]);
    __syncthreads();
  }
  const float m0 = rmin[0];
  for (int n = threadIdx.x; n < NPTS; n += 256) {
    float w2 = 2.f / (1.f + __expf(20.f * (ml[n] - m0)));   // 2*sigmoid(-20*(ml-min))
    wout[(size_t)b * NPTS + n] = (w2 > 0.5f) ? 1.f : 0.f;
  }
}

// ---------------------------------------------------------------------------
// Kernel 4: weighted Procrustes. Two LDS reduction passes (sums, then 3x3
// cov), then thread 0 solves the optimal PROPER rotation via Horn's
// quaternion method (identical to SVD + det-sign fix) with shifted power
// iteration on the 4x4 symmetric matrix.  T = my - R*mx.
// ---------------------------------------------------------------------------
__global__ __launch_bounds__(256) void procrustes_kernel(
    const float* __restrict__ src, const float* __restrict__ tgt,
    const int* __restrict__ corres_i, const float* __restrict__ weight,
    float* __restrict__ outR, float* __restrict__ outT) {
  __shared__ float acc[256][16];
  __shared__ float bc[7];
  const int b = blockIdx.x, t = threadIdx.x;
  const float* sp = src + (size_t)b * 3 * NPTS;
  const float* tp = tgt + (size_t)b * 3 * NPTS;
  const float* wp = weight + (size_t)b * NPTS;
  const int*   ip = corres_i + (size_t)b * NPTS;

  float a[7] = {0.f, 0.f, 0.f, 0.f, 0.f, 0.f, 0.f};
  for (int n = t; n < NPTS; n += 256) {
    float w = fabsf(wp[n]);
    float x0 = sp[n], x1 = sp[NPTS + n], x2 = sp[2 * NPTS + n];
    int idx = ip[n];
    float y0 = tp[idx], y1 = tp[NPTS + idx], y2 = tp[2 * NPTS + idx];
    a[0] += w;
    a[1] += w * x0; a[2] += w * x1; a[3] += w * x2;
    a[4] += w * y0; a[5] += w * y1; a[6] += w * y2;
  }
  for (int q = 0; q < 7; ++q) acc[t][q] = a[q];
  for (int q = 7; q < 16; ++q) acc[t][q] = 0.f;
  __syncthreads();
  for (int st = 128; st > 0; st >>= 1) {
    if (t < st)
      for (int q = 0; q < 16; ++q) acc[t][q] += acc[t + st][q];
    __syncthreads();
  }
  if (t == 0) {
    float inv = 1.f / (acc[0][0] + 1e-7f);
    bc[0] = acc[0][1] * inv; bc[1] = acc[0][2] * inv; bc[2] = acc[0][3] * inv;  // mx
    bc[3] = acc[0][4] * inv; bc[4] = acc[0][5] * inv; bc[5] = acc[0][6] * inv;  // my
    bc[6] = inv;
  }
  __syncthreads();
  const float mx0 = bc[0], mx1 = bc[1], mx2 = bc[2];
  const float my0 = bc[3], my1 = bc[4], my2 = bc[5], inv = bc[6];

  float c[9] = {0.f, 0.f, 0.f, 0.f, 0.f, 0.f, 0.f, 0.f, 0.f};
  for (int n = t; n < NPTS; n += 256) {
    float w = fabsf(wp[n]) * inv;
    float x0 = sp[n] - mx0, x1 = sp[NPTS + n] - mx1, x2 = sp[2 * NPTS + n] - mx2;
    int idx = ip[n];
    float y0 = tp[idx] - my0, y1 = tp[NPTS + idx] - my1, y2 = tp[2 * NPTS + idx] - my2;
    c[0] += y0 * w * x0; c[1] += y0 * w * x1; c[2] += y0 * w * x2;
    c[3] += y1 * w * x0; c[4] += y1 * w * x1; c[5] += y1 * w * x2;
    c[6] += y2 * w * x0; c[7] += y2 * w * x1; c[8] += y2 * w * x2;
  }
  for (int q = 0; q < 9; ++q) acc[t][q] = c[q];
  for (int q = 9; q < 16; ++q) acc[t][q] = 0.f;
  __syncthreads();
  for (int st = 128; st > 0; st >>= 1) {
    if (t < st)
      for (int q = 0; q < 16; ++q) acc[t][q] += acc[t + st][q];
    __syncthreads();
  }
  if (t == 0) {
    // cov[i][j] = sum_n (Y-my)_i * nw * (X-mx)_j ;  Horn's S_ab = cov[b][a]
    float s00 = acc[0][0], s10 = acc[0][1], s20 = acc[0][2];
    float s01 = acc[0][3], s11 = acc[0][4], s21 = acc[0][5];
    float s02 = acc[0][6], s12 = acc[0][7], s22 = acc[0][8];
    float K[4][4];
    K[0][0] = s00 + s11 + s22;
    K[0][1] = K[1][0] = s12 - s21;
    K[0][2] = K[2][0] = s20 - s02;
    K[0][3] = K[3][0] = s01 - s10;
    K[1][1] = s00 - s11 - s22;
    K[1][2] = K[2][1] = s01 + s10;
    K[1][3] = K[3][1] = s20 + s02;
    K[2][2] = -s00 + s11 - s22;
    K[2][3] = K[3][2] = s12 + s21;
    K[3][3] = -s00 - s11 + s22;
    float sh = 1e-6f;
    for (int i = 0; i < 4; ++i)
      for (int j = 0; j < 4; ++j) sh += fabsf(K[i][j]);  // make K+sh*I PSD
    float q0 = 1.f, q1 = 0.01f, q2 = 0.02f, q3 = 0.03f;
    for (int it = 0; it < 100; ++it) {
      float r0 = sh * q0 + K[0][0] * q0 + K[0][1] * q1 + K[0][2] * q2 + K[0][3] * q3;
      float r1 = sh * q1 + K[1][0] * q0 + K[1][1] * q1 + K[1][2] * q2 + K[1][3] * q3;
      float r2 = sh * q2 + K[2][0] * q0 + K[2][1] * q1 + K[2][2] * q2 + K[2][3] * q3;
      float r3 = sh * q3 + K[3][0] * q0 + K[3][1] * q1 + K[3][2] * q2 + K[3][3] * q3;
      float nrm = rsqrtf(r0 * r0 + r1 * r1 + r2 * r2 + r3 * r3 + 1e-30f);
      q0 = r0 * nrm; q1 = r1 * nrm; q2 = r2 * nrm; q3 = r3 * nrm;
    }
    float R[3][3];
    R[0][0] = q0 * q0 + q1 * q1 - q2 * q2 - q3 * q3;
    R[0][1] = 2.f * (q1 * q2 - q0 * q3);
    R[0][2] = 2.f * (q1 * q3 + q0 * q2);
    R[1][0] = 2.f * (q1 * q2 + q0 * q3);
    R[1][1] = q0 * q0 - q1 * q1 + q2 * q2 - q3 * q3;
    R[1][2] = 2.f * (q2 * q3 - q0 * q1);
    R[2][0] = 2.f * (q1 * q3 - q0 * q2);
    R[2][1] = 2.f * (q2 * q3 + q0 * q1);
    R[2][2] = q0 * q0 - q1 * q1 - q2 * q2 + q3 * q3;
    float mx[3] = {mx0, mx1, mx2}, my[3] = {my0, my1, my2};
    for (int i = 0; i < 3; ++i) {
      float ti = my[i];
      for (int j = 0; j < 3; ++j) {
        outR[(size_t)b * 9 + i * 3 + j] = R[i][j];
        ti -= R[i][j] * mx[j];
      }
      outT[(size_t)b * 3 + i] = ti;
    }
  }
}

// ---------------------------------------------------------------------------
// inputs: src_embedding(16*512*1024 f32), tgt_embedding, src(16*3*1024 f32),
//         tgt(16*3*1024 f32), temperature(16, unused), is_corr(1, unused)
// d_out: [R 144][T 48][corres 16384][weight 16384]  (all f32)
// ws:    4x bf16 NxD matrices (64 MB) + corr_tgt + corres + mean_loss (~67.4 MB)
// ---------------------------------------------------------------------------
extern "C" void kernel_launch(void* const* d_in, const int* in_sizes, int n_in,
                              void* d_out, int out_size, void* d_ws, size_t ws_size,
                              hipStream_t stream) {
  (void)in_sizes; (void)n_in; (void)out_size; (void)ws_size;
  const float* src_emb = (const float*)d_in[0];
  const float* tgt_emb = (const float*)d_in[1];
  const float* src     = (const float*)d_in[2];
  const float* tgt     = (const float*)d_in[3];

  float* out = (float*)d_out;
  char*  ws  = (char*)d_ws;
  const size_t EMB = (size_t)BATCH * NPTS * DIM;  // elements per matrix
  unsigned short* sh = (unsigned short*)ws;
  unsigned short* sl = sh + EMB;
  unsigned short* th = sl + EMB;
  unsigned short* tl = th + EMB;
  float* corrT  = (float*)(ws + 4 * EMB * sizeof(unsigned short));
  int*   corres = (int*)(corrT + (size_t)BATCH * NPTS * 3);
  float* mloss  = (float*)(corres + (size_t)BATCH * NPTS);

  float* outR = out;                          // 144
  float* outT = out + 144;                    // 48
  float* outC = out + 192;                    // 16384
  float* outW = out + 192 + BATCH * NPTS;     // 16384

  dim3 tb(32, 8);
  dim3 tg(BATCH, NPTS / 32, DIM / 32);
  split_transpose_kernel<<<tg, tb, 0, stream>>>(src_emb, sh, sl);
  split_transpose_kernel<<<tg, tb, 0, stream>>>(tgt_emb, th, tl);
  attn_kernel<<<BATCH * 64, 128, 0, stream>>>(sh, sl, th, tl, tgt, corrT, corres, outC);
  gfm_kernel<<<BATCH * 4, 256, 0, stream>>>(src, corrT, mloss);
  weight_kernel<<<BATCH, 256, 0, stream>>>(mloss, outW);
  procrustes_kernel<<<BATCH, 256, 0, stream>>>(src, tgt, corres, outW, outR, outT);
}
